// Net_42391327211590
// MI455X (gfx1250) — matrix operand order repaired and checked by reading, not compiled, not run on hardware
//
#include <hip/hip_runtime.h>
#include <hip/hip_bf16.h>
#include <math.h>

// ---- CDNA5 WMMA vector types -------------------------------------------------
typedef __attribute__((ext_vector_type(16))) _Float16 v16h;
typedef __attribute__((ext_vector_type(8)))  _Float16 f16x8;
typedef __attribute__((ext_vector_type(4)))  _Float16 f16x4;
typedef __attribute__((ext_vector_type(8)))  float    v8f;

#define SHUF16 0,1,2,3,4,5,6,7,8,9,10,11,12,13,14,15

__device__ __forceinline__ float selu_f(float x) {
    const float scale = 1.0507009873554805f;
    const float alpha = 1.6732632423543772f;
    return scale * (x > 0.f ? x : alpha * (expf(x) - 1.f));
}

// ---- Prep kernels ------------------------------------------------------------
__global__ void conv_x_f16(const float* __restrict__ x, _Float16* __restrict__ xh, long long n4) {
    long long stride = (long long)gridDim.x * blockDim.x;
    for (long long i = blockIdx.x * (long long)blockDim.x + threadIdx.x; i < n4; i += stride) {
        float4 v = ((const float4*)x)[i];
        f16x4 h = { (_Float16)v.x, (_Float16)v.y, (_Float16)v.z, (_Float16)v.w };
        ((f16x4*)xh)[i] = h;
    }
}

// W1 [512,128] -> W1t [128,512] f16 (B operand, K contiguous per output column)
__global__ void prep_W1t(const float* __restrict__ W1, _Float16* __restrict__ W1t) {
    int idx = blockIdx.x * blockDim.x + threadIdx.x;
    if (idx < 512 * 128) {
        int k = idx >> 7, c = idx & 127;
        W1t[c * 512 + k] = (_Float16)W1[idx];
    }
}

// W2 [128,40] -> W2t [48,128] f16, zero-padded cols 40..47
__global__ void prep_W2t(const float* __restrict__ W2, _Float16* __restrict__ W2t) {
    int idx = blockIdx.x * blockDim.x + threadIdx.x;
    if (idx < 48 * 128) {
        int c = idx >> 7, k = idx & 127;
        W2t[idx] = (c < 40) ? (_Float16)W2[k * 40 + c] : (_Float16)0.0f;
    }
}

// ---- GEMM1: P0 = selu(x @ W1 + b1), M=N rows, K=512, 128 cols ----------------
// one wave = 16-row stripe, 8 column tiles. Per k-step: batch-issue A + all 8 B
// tiles as one load clause, then 8 back-to-back WMMAs (progressive waits).
__global__ __launch_bounds__(256) void gemm1_selu(
    const _Float16* __restrict__ xh, const _Float16* __restrict__ W1t,
    const float* __restrict__ b1, float* __restrict__ P0, int N)
{
    const int lane = threadIdx.x & 31;
    const int wave = blockIdx.x * (blockDim.x >> 5) + (threadIdx.x >> 5);
    const int m0 = wave * 16;
    if (m0 >= N) return;
    const int hi = lane >> 4;       // half-wave select
    const int r  = lane & 15;
    int arow = m0 + r; if (arow >= N) arow = N - 1;
    const _Float16* aptr = xh + (size_t)arow * 512;
    // B 32x16 f16 fragment: lanes0-15 K0-15, lanes16-31 K16-31 (N = lane&15)
    const _Float16* bbase = W1t + (size_t)r * 512 + hi * 16;

    v8f acc[8] = {};
#pragma unroll 2
    for (int kt = 0; kt < 16; ++kt) {
        const int k0 = kt * 32;
        // A 16x32 f16 fragment: lanes0-15 K0-15, lanes16-31 K16-31
        f16x8 alo = *(const f16x8*)(aptr + k0 + hi * 16);
        f16x8 ahi = *(const f16x8*)(aptr + k0 + 8 + hi * 16);
        v16h a = __builtin_shufflevector(alo, ahi, SHUF16);
        v16h b[8];
#pragma unroll
        for (int ct = 0; ct < 8; ++ct)
            b[ct] = *(const v16h*)(bbase + (size_t)ct * (16 * 512) + k0);
#pragma unroll
        for (int ct = 0; ct < 8; ++ct)
            acc[ct] = __builtin_amdgcn_wmma_f32_16x16x32_f16(
                false, a, false, b[ct], (short)0, acc[ct], false, false);
    }
#pragma unroll
    for (int ct = 0; ct < 8; ++ct) {
        const int col = ct * 16 + r;
        const float bias = b1[col];
#pragma unroll
        for (int v = 0; v < 8; ++v) {
            const int m = m0 + v + hi * 8;   // D layout: lanes16-31 hold M=8..15
            if (m < N) P0[(size_t)m * 128 + col] = selu_f(acc[ct][v] + bias);
        }
    }
}

// ---- Degree / dinv -----------------------------------------------------------
__global__ void deg_init(float* __restrict__ deg, int N) {
    for (int i = blockIdx.x * blockDim.x + threadIdx.x; i < N; i += gridDim.x * blockDim.x)
        deg[i] = 1.0f;  // self-loop
}
__global__ void deg_count(const int* __restrict__ col, float* __restrict__ deg, int E) {
    for (int e = blockIdx.x * blockDim.x + threadIdx.x; e < E; e += gridDim.x * blockDim.x)
        atomicAdd(&deg[col[e]], 1.0f);
}
__global__ void deg_rsqrt(float* __restrict__ deg, int N) {
    for (int i = blockIdx.x * blockDim.x + threadIdx.x; i < N; i += gridDim.x * blockDim.x) {
        float d = deg[i];
        deg[i] = (d > 0.f) ? rsqrtf(d) : 0.f;
    }
}

// ---- Propagation: Pout = Â Pin ----------------------------------------------
__global__ void prop_self(const float* __restrict__ dinv, const float* __restrict__ Pin,
                          float* __restrict__ Pout, int N) {
    long long total = (long long)N * 32;
    long long stride = (long long)gridDim.x * blockDim.x;
    for (long long i = blockIdx.x * (long long)blockDim.x + threadIdx.x; i < total; i += stride) {
        int n = (int)(i >> 5), c = (int)(i & 31);
        float d = dinv[n], s = d * d;
        float4 v = ((const float4*)(Pin + (size_t)n * 128))[c];
        float4 o; o.x = s * v.x; o.y = s * v.y; o.z = s * v.z; o.w = s * v.w;
        ((float4*)(Pout + (size_t)n * 128))[c] = o;
    }
}

// wave per edge: 32 lanes x float4 = 128 features gather + atomic scatter
__global__ void prop_edges(const int* __restrict__ ei, const float* __restrict__ dinv,
                           const float* __restrict__ Pin, float* __restrict__ Pout, int E) {
    const int lane = threadIdx.x & 31;
    int wave = (blockIdx.x * blockDim.x + threadIdx.x) >> 5;
    int nw = (gridDim.x * blockDim.x) >> 5;
    const int* rowp = ei;
    const int* colp = ei + E;
    for (int e = wave; e < E; e += nw) {
        int r = rowp[e], c = colp[e];
        float nm = dinv[r] * dinv[c];
        float4 v = *(const float4*)(Pin + (size_t)r * 128 + lane * 4);
        float* dst = Pout + (size_t)c * 128 + lane * 4;
        atomicAdd(dst + 0, nm * v.x);
        atomicAdd(dst + 1, nm * v.y);
        atomicAdd(dst + 2, nm * v.z);
        atomicAdd(dst + 3, nm * v.w);
    }
}

// ---- fp[n,k] = P_k[n]·proj_w + proj_b ; S[n,:] (+)= fp * P_k[n,:] ------------
__global__ void fp_and_S(const float* __restrict__ Pk, const float* __restrict__ proj_w,
                         const float* __restrict__ proj_b, float* __restrict__ fp,
                         float* __restrict__ S, int N, int k, int first) {
    const int lane = threadIdx.x & 31;
    int wave = (blockIdx.x * blockDim.x + threadIdx.x) >> 5;
    int nw = (gridDim.x * blockDim.x) >> 5;
    float4 w = *(const float4*)(proj_w + lane * 4);
    float pb = proj_b[0];
    for (int n = wave; n < N; n += nw) {
        float4 v = *(const float4*)(Pk + (size_t)n * 128 + lane * 4);
        float p = v.x * w.x + v.y * w.y + v.z * w.z + v.w * w.w;
        for (int off = 16; off > 0; off >>= 1) p += __shfl_xor(p, off, 32);
        float f = p + pb;
        if (lane == 0) fp[(size_t)n * 12 + k] = f;
        float* s = S + (size_t)n * 128 + lane * 4;
        if (first) {
            s[0] = f * v.x; s[1] = f * v.y; s[2] = f * v.z; s[3] = f * v.w;
        } else {
            s[0] += f * v.x; s[1] += f * v.y; s[2] += f * v.z; s[3] += f * v.w;
        }
    }
}

// ---- mean over N of fp -------------------------------------------------------
__global__ void zero_small(float* __restrict__ p, int n) {
    int i = blockIdx.x * blockDim.x + threadIdx.x;
    if (i < n) p[i] = 0.f;
}
__global__ void reduce_m(const float* __restrict__ fp, float* __restrict__ msum, int N) {
    __shared__ float sd[256];
    float acc[11];
#pragma unroll
    for (int k = 0; k < 11; ++k) acc[k] = 0.f;
    for (int n = blockIdx.x * blockDim.x + threadIdx.x; n < N; n += gridDim.x * blockDim.x)
#pragma unroll
        for (int k = 0; k < 11; ++k) acc[k] += fp[(size_t)n * 12 + k];
    for (int k = 0; k < 11; ++k) {
        sd[threadIdx.x] = acc[k];
        __syncthreads();
        for (int s = 128; s > 0; s >>= 1) {
            if ((int)threadIdx.x < s) sd[threadIdx.x] += sd[threadIdx.x + s];
            __syncthreads();
        }
        if (threadIdx.x == 0) atomicAdd(&msum[k], sd[0]);
        __syncthreads();
    }
}

// ---- share_w = l2norm(mean + hop_w) ------------------------------------------
__global__ void share_w_kernel(const float* __restrict__ msum, const float* __restrict__ hop_w,
                               float* __restrict__ sw, int N) {
    int l = threadIdx.x;
    float v = (l < 11) ? (msum[l] / (float)N + hop_w[l]) : 0.f;
    float ss = v * v;
    for (int off = 16; off > 0; off >>= 1) ss += __shfl_xor(ss, off, 32);
    float nrm = fmaxf(sqrtf(ss), 1e-12f);
    if (l < 11) sw[l] = v / nrm;
}

// ---- pooling + tanh -> h2h (f16) ---------------------------------------------
__global__ void pool_tanh(const float* __restrict__ P, const float* __restrict__ S,
                          const float* __restrict__ fp, const float* __restrict__ sw,
                          _Float16* __restrict__ h2h, int N) {
    const int lane = threadIdx.x & 31;
    int wave = (blockIdx.x * blockDim.x + threadIdx.x) >> 5;
    int nw = (gridDim.x * blockDim.x) >> 5;
    const size_t Pstride = (size_t)N * 128;
    float swv = (lane < 11) ? sw[lane] : 0.f;
    for (int n = wave; n < N; n += nw) {
        float f = (lane < 11) ? fp[(size_t)n * 12 + lane] : 0.f;
        float ss = f * f;
        for (int off = 16; off > 0; off >>= 1) ss += __shfl_xor(ss, off, 32);
        float inv = 0.8f / fmaxf(sqrtf(ss), 1e-12f);   // fold 0.8*selfr
        float ax = 0.f, ay = 0.f, az = 0.f, aw = 0.f;
#pragma unroll
        for (int k = 0; k < 11; ++k) {
            float wk = 0.2f * __shfl(swv, k, 32);      // fold 0.2*share
            float4 p = *(const float4*)(P + (size_t)k * Pstride + (size_t)n * 128 + lane * 4);
            ax += wk * p.x; ay += wk * p.y; az += wk * p.z; aw += wk * p.w;
        }
        float4 s4 = *(const float4*)(S + (size_t)n * 128 + lane * 4);
        f16x4 h = { (_Float16)tanhf(ax + inv * s4.x), (_Float16)tanhf(ay + inv * s4.y),
                    (_Float16)tanhf(az + inv * s4.z), (_Float16)tanhf(aw + inv * s4.w) };
        *(f16x4*)(h2h + (size_t)n * 128 + lane * 4) = h;
    }
}

// ---- GEMM2: logits[N,48] = h2h @ W2 + b2 (cols>=40 padded) -------------------
__global__ __launch_bounds__(256) void gemm2_k(
    const _Float16* __restrict__ h2h, const _Float16* __restrict__ W2t,
    const float* __restrict__ b2, float* __restrict__ logits, int N)
{
    const int lane = threadIdx.x & 31;
    const int wave = blockIdx.x * (blockDim.x >> 5) + (threadIdx.x >> 5);
    const int m0 = wave * 16;
    if (m0 >= N) return;
    const int hi = lane >> 4;
    const int r  = lane & 15;
    int arow = m0 + r; if (arow >= N) arow = N - 1;
    const _Float16* aptr = h2h + (size_t)arow * 128;
    const _Float16* bbase = W2t + (size_t)r * 128 + hi * 16;

    v8f acc[3] = {};
#pragma unroll
    for (int kt = 0; kt < 4; ++kt) {
        const int k0 = kt * 32;
        f16x8 alo = *(const f16x8*)(aptr + k0 + hi * 16);
        f16x8 ahi = *(const f16x8*)(aptr + k0 + 8 + hi * 16);
        v16h a = __builtin_shufflevector(alo, ahi, SHUF16);
        v16h b[3];
#pragma unroll
        for (int ct = 0; ct < 3; ++ct)
            b[ct] = *(const v16h*)(bbase + (size_t)ct * (16 * 128) + k0);
#pragma unroll
        for (int ct = 0; ct < 3; ++ct)
            acc[ct] = __builtin_amdgcn_wmma_f32_16x16x32_f16(
                false, a, false, b[ct], (short)0, acc[ct], false, false);
    }
#pragma unroll
    for (int ct = 0; ct < 3; ++ct) {
        const int col = ct * 16 + r;
        const float bias = (col < 40) ? b2[col] : 0.f;
#pragma unroll
        for (int v = 0; v < 8; ++v) {
            const int m = m0 + v + hi * 8;
            if (m < N) logits[(size_t)m * 48 + col] = acc[ct][v] + bias;
        }
    }
}

// ---- per-row log_softmax over 40 classes -------------------------------------
__global__ void logsoftmax_k(const float* __restrict__ logits, float* __restrict__ out, int N) {
    const int lane = threadIdx.x & 31;
    int wave = (blockIdx.x * blockDim.x + threadIdx.x) >> 5;
    int nw = (gridDim.x * blockDim.x) >> 5;
    for (int n = wave; n < N; n += nw) {
        const float* r = logits + (size_t)n * 48;
        float x1 = r[lane];
        float x2 = (lane < 8) ? r[32 + lane] : -1e30f;
        float mx = fmaxf(x1, x2);
        for (int off = 16; off > 0; off >>= 1) mx = fmaxf(mx, __shfl_xor(mx, off, 32));
        float s = expf(x1 - mx) + ((lane < 8) ? expf(x2 - mx) : 0.f);
        for (int off = 16; off > 0; off >>= 1) s += __shfl_xor(s, off, 32);
        float lse = mx + logf(s);
        float* o = out + (size_t)n * 40;
        o[lane] = x1 - lse;
        if (lane < 8) o[32 + lane] = x2 - lse;
    }
}

// ---- launcher ----------------------------------------------------------------
extern "C" void kernel_launch(void* const* d_in, const int* in_sizes, int n_in,
                              void* d_out, int out_size, void* d_ws, size_t ws_size,
                              hipStream_t stream) {
    (void)n_in; (void)out_size; (void)ws_size;
    const float* x      = (const float*)d_in[0];
    const int*   ei     = (const int*)d_in[1];
    const float* W1     = (const float*)d_in[2];
    const float* b1     = (const float*)d_in[3];
    const float* proj_w = (const float*)d_in[4];
    const float* proj_b = (const float*)d_in[5];
    const float* hop_w  = (const float*)d_in[6];
    const float* W2     = (const float*)d_in[7];
    const float* b2     = (const float*)d_in[8];
    float* out = (float*)d_out;

    const int N = in_sizes[0] / 512;
    const int E = in_sizes[1] / 2;

    // workspace layout (256B aligned slabs)
    char* w = (char*)d_ws;
    auto alloc = [&](size_t bytes) -> char* {
        char* p = w; w += (bytes + 255) & ~(size_t)255; return p;
    };
    _Float16* xh  = (_Float16*)alloc((size_t)N * 512 * 2);
    _Float16* W1t = (_Float16*)alloc((size_t)128 * 512 * 2);
    _Float16* W2t = (_Float16*)alloc((size_t)48 * 128 * 2);
    float* dinv   = (float*)alloc((size_t)N * 4);
    float* P      = (float*)alloc((size_t)11 * N * 128 * 4);  // P0..P10
    float* S      = (float*)alloc((size_t)N * 128 * 4);
    float* fp     = (float*)alloc((size_t)N * 12 * 4);        // stride 12
    float* msum   = (float*)alloc(256);
    float* sw     = (float*)alloc(256);
    _Float16* h2h = (_Float16*)alloc((size_t)N * 128 * 2);
    float* logits = (float*)alloc((size_t)N * 48 * 4);

    const size_t Pstride = (size_t)N * 128;
    const int gemmBlocks = ((N + 15) / 16 + 7) / 8;  // 8 waves/block, 16 rows/wave

    conv_x_f16<<<4096, 256, 0, stream>>>(x, xh, (long long)N * 128);
    prep_W1t<<<(512 * 128 + 255) / 256, 256, 0, stream>>>(W1, W1t);
    prep_W2t<<<(48 * 128 + 255) / 256, 256, 0, stream>>>(W2, W2t);

    gemm1_selu<<<gemmBlocks, 256, 0, stream>>>(xh, W1t, b1, P, N);

    deg_init<<<1024, 256, 0, stream>>>(dinv, N);
    deg_count<<<4096, 256, 0, stream>>>(ei + E, dinv, E);
    deg_rsqrt<<<1024, 256, 0, stream>>>(dinv, N);

    for (int k = 1; k <= 10; ++k) {
        const float* Pin = P + (size_t)(k - 1) * Pstride;
        float* Pout = P + (size_t)k * Pstride;
        prop_self<<<4096, 256, 0, stream>>>(dinv, Pin, Pout, N);
        prop_edges<<<4096, 256, 0, stream>>>(ei, dinv, Pin, Pout, E);
    }

    for (int k = 0; k <= 10; ++k)
        fp_and_S<<<2048, 256, 0, stream>>>(P + (size_t)k * Pstride, proj_w, proj_b,
                                           fp, S, N, k, (k == 0) ? 1 : 0);

    zero_small<<<1, 32, 0, stream>>>(msum, 11);
    reduce_m<<<256, 256, 0, stream>>>(fp, msum, N);
    share_w_kernel<<<1, 32, 0, stream>>>(msum, hop_w, sw, N);
    pool_tanh<<<2048, 256, 0, stream>>>(P, S, fp, sw, h2h, N);

    gemm2_k<<<gemmBlocks, 256, 0, stream>>>(h2h, W2t, b2, logits, N);
    logsoftmax_k<<<2048, 256, 0, stream>>>(logits, out, N);
}
